// MossFormerBlock_23940147707984
// MI455X (gfx1250) — matrix operand order repaired and checked by reading, not compiled, not run on hardware
//
#include <hip/hip_runtime.h>
#include <hip/hip_bf16.h>
#include <math.h>

// ---------------------------------------------------------------------------
// MossFormer block (FLASH gated attention) for MI455X / gfx1250.
// All GEMMs use v_wmma_f32_16x16x32_bf16 (bf16 in, f32 accumulate).
// Every GEMM consumes A as [M,K] row-major and B as [N,K] row-major, so
// global->LDS staging is pure 16-byte vector copies (b128 loads/stores).
// ---------------------------------------------------------------------------

typedef __bf16 bf16_t;
typedef __attribute__((ext_vector_type(16))) __bf16 v16bf;
typedef __attribute__((ext_vector_type(8)))  float  v8f;
typedef __attribute__((ext_vector_type(4)))  unsigned int u32x4;

#define BB 4
#define TT 4096
#define DD 512
#define HH 1024
#define QKD 128
#define GG 256
#define MM (BB * TT)           // 16384 rows
#define NGRP (BB * (TT / GG))  // 64 groups total

// ---------------------------------------------------------------------------
// WMMA GEMM:  out = epilogue( alpha * A @ B^T_layout )
//   A: bf16 [M,K] row-major (lda = row stride in elements)
//   B: bf16 [N,K] row-major (ldb = row stride in elements)
//   B batch offset = (z / zDiv) * bStrideHi + (z % zDiv) * bStrideLo
//   EPI: 0 = store f32
//        1 = +bias, SiLU, store f32
//        2 = relu(x)^2, store bf16
//        4 = accumulate-add into f32 out
//        5 = store bf16 TRANSPOSED: out[col*ldo + row]
// Block tile 128x64, BK=32, 256 threads = 8 waves (4x2), 32x32 per wave.
// Requires M%128==0, N%64==0, K%32==0; lda/ldb multiples of 8; 16B-aligned
// base pointers (all true for every call below).
// ---------------------------------------------------------------------------
template <int EPI>
__global__ __launch_bounds__(256)
void wmma_gemm(const bf16_t* __restrict__ A,
               const bf16_t* __restrict__ Bsrc,
               const float* __restrict__ bias,
               float* __restrict__ outF,
               bf16_t* __restrict__ outB,
               int M, int N, int K,
               int lda, int ldb, int ldo,
               long long aStride,
               long long bStrideHi, long long bStrideLo, int zDiv,
               long long oStride,
               float alpha) {
  __shared__ bf16_t sA[2][128 * 32];
  __shared__ bf16_t sB[2][64 * 32];

  const int t    = threadIdx.x;
  const int lane = t & 31;
  const int wave = t >> 5;
  const int wm   = (wave >> 1) * 32;   // 0,32,64,96
  const int wn   = (wave & 1) * 32;    // 0,32
  const int m0   = blockIdx.y * 128;
  const int n0   = blockIdx.x * 64;

  const long long zb = blockIdx.z;
  const bf16_t* Ab = A    + zb * aStride;
  const bf16_t* Bb = Bsrc + (zb / zDiv) * bStrideHi + (zb % zDiv) * bStrideLo;

  // 16B-chunk staging: A tile 128x32 -> 512 chunks (2/thread),
  //                    B tile  64x32 -> 256 chunks (1/thread).
  const int ar0 = (t + 0)   >> 2, ac0 = ((t + 0)   & 3) * 8;
  const int ar1 = (t + 256) >> 2, ac1 = ((t + 256) & 3) * 8;
  const int br  = t >> 2,         bc  = (t & 3) * 8;

  auto stage = [&](int buf, int k0) {
    *(u32x4*)&sA[buf][ar0 * 32 + ac0] =
        *(const u32x4*)&Ab[(long long)(m0 + ar0) * lda + (k0 + ac0)];
    *(u32x4*)&sA[buf][ar1 * 32 + ac1] =
        *(const u32x4*)&Ab[(long long)(m0 + ar1) * lda + (k0 + ac1)];
    *(u32x4*)&sB[buf][br * 32 + bc] =
        *(const u32x4*)&Bb[(long long)(n0 + br) * ldb + (k0 + bc)];
  };

  v8f acc[2][2];
#pragma unroll
  for (int i = 0; i < 2; ++i)
#pragma unroll
    for (int j = 0; j < 2; ++j)
      acc[i][j] = (v8f){0.f, 0.f, 0.f, 0.f, 0.f, 0.f, 0.f, 0.f};

  stage(0, 0);
  int buf = 0;
  const int mrow = lane & 15;
  const int hf   = lane >> 4;

  for (int k0 = 0; k0 < K; k0 += 32) {
    __syncthreads();
    if (k0 + 32 < K) {
      stage(buf ^ 1, k0 + 32);
      if (k0 + 64 < K)  // hint next-next A tile (global_prefetch_b8)
        __builtin_prefetch(&Ab[(long long)(m0 + (t >> 1)) * lda + (k0 + 64)], 0, 1);
    }

    // ---- fragments per gfx1250 VGPR layouts ----
    v16bf aF[2], bF[2];
#pragma unroll
    for (int ti = 0; ti < 2; ++ti) {
      const bf16_t* base = &sA[buf][(wm + ti * 16 + mrow) * 32];
#pragma unroll
      for (int p = 0; p < 8; ++p) {
        int kk = ((p < 4) ? 0 : 16) + hf * 8 + (p & 3) * 2;
        aF[ti][2 * p]     = base[kk];
        aF[ti][2 * p + 1] = base[kk + 1];
      }
    }
#pragma unroll
    for (int tj = 0; tj < 2; ++tj) {
      const bf16_t* base = &sB[buf][(wn + tj * 16 + mrow) * 32];
#pragma unroll
      for (int p = 0; p < 8; ++p) {
        int kk = ((p < 4) ? 0 : 16) + hf * 8 + (p & 3) * 2;
        bF[tj][2 * p]     = base[kk];
        bF[tj][2 * p + 1] = base[kk + 1];
      }
    }

#pragma unroll
    for (int ti = 0; ti < 2; ++ti)
#pragma unroll
      for (int tj = 0; tj < 2; ++tj)
        acc[ti][tj] = __builtin_amdgcn_wmma_f32_16x16x32_bf16(
            false, aF[ti], false, bF[tj], (short)0, acc[ti][tj], false, false);

    buf ^= 1;
  }

  // ---- epilogue: C/D layout -> lane L, vgpr r: M = r + (L>=16)*8, N = L&15 ----
  float*  oF = (EPI == 0 || EPI == 1 || EPI == 4) ? outF + zb * oStride : nullptr;
  bf16_t* oB = (EPI == 2 || EPI == 5)             ? outB + zb * oStride : nullptr;
  const int hf2  = lane >> 4;
  const int col4 = lane & 15;
#pragma unroll
  for (int ti = 0; ti < 2; ++ti) {
#pragma unroll
    for (int tj = 0; tj < 2; ++tj) {
#pragma unroll
      for (int r = 0; r < 8; ++r) {
        int row = m0 + wm + ti * 16 + hf2 * 8 + r;
        int col = n0 + wn + tj * 16 + col4;
        float val = acc[ti][tj][r] * alpha;
        if (EPI == 0) {
          oF[(long long)row * ldo + col] = val;
        } else if (EPI == 1) {
          float x = val + bias[col];
          oF[(long long)row * ldo + col] = x / (1.f + __expf(-x));  // SiLU
        } else if (EPI == 2) {
          float rl = val > 0.f ? val : 0.f;
          oB[(long long)row * ldo + col] = (bf16_t)(rl * rl);       // relu^2
        } else if (EPI == 4) {
          oF[(long long)row * ldo + col] += val;
        } else if (EPI == 5) {
          oB[(long long)col * ldo + row] = (bf16_t)val;             // transposed
        }
      }
    }
  }
}

// ---------------------------------------------------------------------------
// LayerNorm row statistics: mu, rstd per row.
// ---------------------------------------------------------------------------
__global__ __launch_bounds__(256)
void ln_stats(const float* __restrict__ x, float* __restrict__ stats, int C) {
  __shared__ float sh[512];
  int row = blockIdx.x;
  const float* xr = x + (long long)row * C;
  float s = 0.f, s2 = 0.f;
  for (int c = threadIdx.x; c < C; c += 256) {
    float v = xr[c];
    s += v; s2 += v * v;
  }
  sh[threadIdx.x] = s;
  sh[threadIdx.x + 256] = s2;
  __syncthreads();
  for (int off = 128; off > 0; off >>= 1) {
    if (threadIdx.x < off) {
      sh[threadIdx.x]       += sh[threadIdx.x + off];
      sh[256 + threadIdx.x] += sh[256 + threadIdx.x + off];
    }
    __syncthreads();
  }
  if (threadIdx.x == 0) {
    float mu  = sh[0] / (float)C;
    float var = sh[256] / (float)C - mu * mu;
    stats[2 * row]     = mu;
    stats[2 * row + 1] = rsqrtf(var + 1e-5f);
  }
}

// xn = (x - mu) * rstd * g + b   (store bf16)
__global__ __launch_bounds__(256)
void ln_apply(const float* __restrict__ x, const float* __restrict__ stats,
              const float* __restrict__ g, const float* __restrict__ b,
              bf16_t* __restrict__ out, int C, long long total) {
  long long idx = (long long)blockIdx.x * 256 + threadIdx.x;
  if (idx >= total) return;
  long long row = idx / C;
  int c = (int)(idx - row * C);
  float mu = stats[2 * row], rs = stats[2 * row + 1];
  out[idx] = (bf16_t)((x[idx] - mu) * rs * g[c] + b[c]);
}

// depthwise conv (k=3, pad=1) residual:  out = h + conv(h) [+ res]
__global__ __launch_bounds__(256)
void dwconv(const float* __restrict__ h, const float* __restrict__ cw,
            const float* __restrict__ cb, const float* __restrict__ res,
            float* __restrict__ out, int T, int C, long long total) {
  long long idx = (long long)blockIdx.x * 256 + threadIdx.x;
  if (idx >= total) return;
  int c = (int)(idx % C);
  int t = (int)((idx / C) % T);
  float h0 = h[idx];
  float hm = (t > 0)     ? h[idx - C] : 0.f;
  float hp = (t < T - 1) ? h[idx + C] : 0.f;
  float conv = hm * cw[c * 3 + 0] + h0 * cw[c * 3 + 1] + hp * cw[c * 3 + 2] + cb[c];
  float v = h0 + conv;
  if (res) v += res[idx];
  out[idx] = v;
}

// OffsetScale (4 heads) + interleaved rotary; bf16 heads.
// Also writes lin_k transposed: lkT[b][c][t]  ([B, QKD, T]).
__global__ __launch_bounds__(256)
void offset_rotary(const float* __restrict__ z, const float* __restrict__ gamma,
                   const float* __restrict__ beta,
                   bf16_t* __restrict__ qq, bf16_t* __restrict__ lq,
                   bf16_t* __restrict__ qk, bf16_t* __restrict__ lk,
                   bf16_t* __restrict__ lkT,
                   int T, long long total /* rows*64 */) {
  long long idx = (long long)blockIdx.x * 256 + threadIdx.x;
  if (idx >= total) return;
  int i = (int)(idx & 63);                 // pair index 0..63
  long long row = idx >> 6;                // global (b*T + t)
  int t = (int)(row % T);
  long long b = row / T;
  int c0 = 2 * i, c1 = 2 * i + 1;
  float z0 = z[row * QKD + c0];
  float z1 = z[row * QKD + c1];
  float invf = __expf(-logf(10000.f) * (float)(2 * i) / (float)QKD);
  float ang = (float)t * invf;
  float cs = __cosf(ang), sn = __sinf(ang);
  bf16_t* outs[4] = {qq, lq, qk, lk};
#pragma unroll
  for (int hd = 0; hd < 4; ++hd) {
    float a0 = z0 * gamma[hd * QKD + c0] + beta[hd * QKD + c0];
    float a1 = z1 * gamma[hd * QKD + c1] + beta[hd * QKD + c1];
    bf16_t r0 = (bf16_t)(a0 * cs - a1 * sn);
    bf16_t r1 = (bf16_t)(a1 * cs + a0 * sn);
    outs[hd][row * QKD + c0] = r0;
    outs[hd][row * QKD + c1] = r1;
    if (hd == 3) {
      lkT[(b * QKD + c0) * T + t] = r0;
      lkT[(b * QKD + c1) * T + t] = r1;
    }
  }
}

// f32 [B*T, C] -> bf16 transposed [B, C, T]
__global__ __launch_bounds__(256)
void f2bT(const float* __restrict__ in, bf16_t* __restrict__ out,
          int T, int C, long long total /* B*T*C */) {
  long long idx = (long long)blockIdx.x * 256 + threadIdx.x;
  if (idx >= total) return;
  long long ct = (long long)C * T;
  long long b = idx / ct;
  long long rem = idx - b * ct;
  int c = (int)(rem / T);
  int t = (int)(rem - (long long)c * T);
  out[idx] = (bf16_t)in[(b * T + t) * (long long)C + c];
}

// weight [K,N] f32 -> transposed bf16 [N,K]
__global__ __launch_bounds__(256)
void wt_convert(const float* __restrict__ w, bf16_t* __restrict__ wt,
                int K, int N, long long total /* N*K */) {
  long long idx = (long long)blockIdx.x * 256 + threadIdx.x;
  if (idx >= total) return;
  long long n = idx / K;
  int k = (int)(idx - n * K);
  wt[idx] = (bf16_t)w[(long long)k * N + n];
}

// gated = u_hat * v * sigmoid(u * v_hat)
__global__ __launch_bounds__(256)
void gating(const float* __restrict__ u, const float* __restrict__ v,
            const float* __restrict__ u_hat, const float* __restrict__ v_hat,
            float* __restrict__ gated, long long total) {
  long long idx = (long long)blockIdx.x * 256 + threadIdx.x;
  if (idx >= total) return;
  float og = 1.f / (1.f + __expf(-(u[idx] * v_hat[idx])));
  gated[idx] = u_hat[idx] * v[idx] * og;
}

// ---------------------------------------------------------------------------
// Host launch
// ---------------------------------------------------------------------------
static inline size_t al256(size_t x) { return (x + 255) & ~(size_t)255; }
static inline int grid1d(long long total) { return (int)((total + 255) / 256); }

extern "C" void kernel_launch(void* const* d_in, const int* in_sizes, int n_in,
                              void* d_out, int out_size, void* d_ws, size_t ws_size,
                              hipStream_t stream) {
  const float* x       = (const float*)d_in[0];
  const float* u_ln_g  = (const float*)d_in[1];
  const float* u_ln_b  = (const float*)d_in[2];
  const float* u_w     = (const float*)d_in[3];
  const float* u_b     = (const float*)d_in[4];
  const float* u_cw    = (const float*)d_in[5];
  const float* u_cb    = (const float*)d_in[6];
  const float* v_ln_g  = (const float*)d_in[7];
  const float* v_ln_b  = (const float*)d_in[8];
  const float* v_w     = (const float*)d_in[9];
  const float* v_b     = (const float*)d_in[10];
  const float* v_cw    = (const float*)d_in[11];
  const float* v_cb    = (const float*)d_in[12];
  const float* qk_ln_g = (const float*)d_in[13];
  const float* qk_ln_b = (const float*)d_in[14];
  const float* qk_w    = (const float*)d_in[15];
  const float* qk_b    = (const float*)d_in[16];
  const float* qk_cw   = (const float*)d_in[17];
  const float* qk_cb   = (const float*)d_in[18];
  const float* o_ln_g  = (const float*)d_in[19];
  const float* o_ln_b  = (const float*)d_in[20];
  const float* o_w     = (const float*)d_in[21];
  const float* o_b     = (const float*)d_in[22];
  const float* o_cw    = (const float*)d_in[23];
  const float* o_cb    = (const float*)d_in[24];
  const float* gamma   = (const float*)d_in[25];
  const float* beta    = (const float*)d_in[26];
  float* out = (float*)d_out;

  // ---- workspace layout (256B aligned slabs, reused where lifetimes allow) ----
  char* w = (char*)d_ws;
  size_t off = 0;
  auto alloc = [&](size_t bytes) -> void* {
    void* p = w + off;
    off = al256(off + bytes);
    return p;
  };
  bf16_t* Wt_u    = (bf16_t*)alloc((size_t)HH * DD * 2);
  bf16_t* Wt_v    = (bf16_t*)alloc((size_t)HH * DD * 2);
  bf16_t* Wt_qk   = (bf16_t*)alloc((size_t)QKD * DD * 2);
  bf16_t* Wt_o    = (bf16_t*)alloc((size_t)DD * HH * 2);
  float*  stats   = (float*)alloc((size_t)MM * 2 * 4);
  bf16_t* xn      = (bf16_t*)alloc((size_t)MM * DD * 2);
  float*  h_tmp   = (float*)alloc((size_t)MM * HH * 4);   // reused as gated
  float*  u_f     = (float*)alloc((size_t)MM * HH * 4);
  float*  v_f     = (float*)alloc((size_t)MM * HH * 4);
  float*  z_f     = (float*)alloc((size_t)MM * QKD * 4);
  bf16_t* qqh     = (bf16_t*)alloc((size_t)MM * QKD * 2);
  bf16_t* lqh     = (bf16_t*)alloc((size_t)MM * QKD * 2);
  bf16_t* qkh     = (bf16_t*)alloc((size_t)MM * QKD * 2);
  bf16_t* lkh     = (bf16_t*)alloc((size_t)MM * QKD * 2);
  bf16_t* lkT     = (bf16_t*)alloc((size_t)BB * QKD * TT * 2);
  bf16_t* u_bfT   = (bf16_t*)alloc((size_t)BB * HH * TT * 2);  // reused as xn_out
  bf16_t* v_bfT   = (bf16_t*)alloc((size_t)BB * HH * TT * 2);  // reused as h_out f32
  bf16_t* lin_kvT = (bf16_t*)alloc((size_t)BB * HH * QKD * 2);
  bf16_t* lin_kuT = (bf16_t*)alloc((size_t)BB * HH * QKD * 2);
  bf16_t* attn    = (bf16_t*)alloc((size_t)NGRP * GG * GG * 2);
  float*  v_hat   = (float*)alloc((size_t)MM * HH * 4);
  float*  u_hat   = (float*)alloc((size_t)MM * HH * 4);
  bf16_t* xn_out  = u_bfT;          // reuse (dead after lin_ku GEMM)
  float*  h_out   = (float*)v_bfT;  // reuse (dead after lin_kv GEMM)

  const dim3 blk(256);

  // 1) weights -> transposed bf16 [N,K]
  wt_convert<<<grid1d((long long)HH * DD), blk, 0, stream>>>(u_w, Wt_u, DD, HH, (long long)HH * DD);
  wt_convert<<<grid1d((long long)HH * DD), blk, 0, stream>>>(v_w, Wt_v, DD, HH, (long long)HH * DD);
  wt_convert<<<grid1d((long long)QKD * DD), blk, 0, stream>>>(qk_w, Wt_qk, DD, QKD, (long long)QKD * DD);
  wt_convert<<<grid1d((long long)DD * HH), blk, 0, stream>>>(o_w, Wt_o, HH, DD, (long long)DD * HH);

  // 2) LN stats of x (shared by u/v/qk branches)
  ln_stats<<<MM, blk, 0, stream>>>(x, stats, DD);

  // --- u branch: LN -> GEMM(+bias,SiLU) -> dwconv residual ---
  ln_apply<<<grid1d((long long)MM * DD), blk, 0, stream>>>(x, stats, u_ln_g, u_ln_b, xn, DD, (long long)MM * DD);
  wmma_gemm<1><<<dim3(HH / 64, MM / 128, 1), blk, 0, stream>>>(
      xn, Wt_u, u_b, h_tmp, nullptr, MM, HH, DD, DD, DD, HH, 0, 0, 0, 1, 0, 1.f);
  dwconv<<<grid1d((long long)MM * HH), blk, 0, stream>>>(h_tmp, u_cw, u_cb, nullptr, u_f, TT, HH, (long long)MM * HH);

  // --- v branch ---
  ln_apply<<<grid1d((long long)MM * DD), blk, 0, stream>>>(x, stats, v_ln_g, v_ln_b, xn, DD, (long long)MM * DD);
  wmma_gemm<1><<<dim3(HH / 64, MM / 128, 1), blk, 0, stream>>>(
      xn, Wt_v, v_b, h_tmp, nullptr, MM, HH, DD, DD, DD, HH, 0, 0, 0, 1, 0, 1.f);
  dwconv<<<grid1d((long long)MM * HH), blk, 0, stream>>>(h_tmp, v_cw, v_cb, nullptr, v_f, TT, HH, (long long)MM * HH);

  // --- qk branch ---
  ln_apply<<<grid1d((long long)MM * DD), blk, 0, stream>>>(x, stats, qk_ln_g, qk_ln_b, xn, DD, (long long)MM * DD);
  wmma_gemm<1><<<dim3(QKD / 64, MM / 128, 1), blk, 0, stream>>>(
      xn, Wt_qk, qk_b, h_tmp, nullptr, MM, QKD, DD, DD, DD, QKD, 0, 0, 0, 1, 0, 1.f);
  dwconv<<<grid1d((long long)MM * QKD), blk, 0, stream>>>(h_tmp, qk_cw, qk_cb, nullptr, z_f, TT, QKD, (long long)MM * QKD);

  // 3) OffsetScale + rotary -> 4 bf16 heads (+ transposed lin_k)
  offset_rotary<<<grid1d((long long)MM * (QKD / 2)), blk, 0, stream>>>(
      z_f, gamma, beta, qqh, lqh, qkh, lkh, lkT, TT, (long long)MM * (QKD / 2));

  // 4) transposed bf16 copies of u, v:  [B, H, T]
  f2bT<<<grid1d((long long)MM * HH), blk, 0, stream>>>(u_f, u_bfT, TT, HH, (long long)MM * HH);
  f2bT<<<grid1d((long long)MM * HH), blk, 0, stream>>>(v_f, v_bfT, TT, HH, (long long)MM * HH);

  // 5) quadratic scores: attn = relu(qq @ qk^T / G)^2   per (batch,group)
  wmma_gemm<2><<<dim3(GG / 64, GG / 128, NGRP), blk, 0, stream>>>(
      qqh, qkh, nullptr, nullptr, attn, GG, GG, QKD, QKD, QKD, GG,
      (long long)GG * QKD, (long long)GG * QKD, 0, 1, (long long)GG * GG, 1.f / (float)GG);

  // 6) quad_out_v / quad_out_u: attn @ v (B from transposed v: [n][b*H*T + g*G + k])
  wmma_gemm<0><<<dim3(HH / 64, GG / 128, NGRP), blk, 0, stream>>>(
      attn, v_bfT, nullptr, v_hat, nullptr, GG, HH, GG, GG, TT, HH,
      (long long)GG * GG, (long long)HH * TT, GG, 16, (long long)GG * HH, 1.f);
  wmma_gemm<0><<<dim3(HH / 64, GG / 128, NGRP), blk, 0, stream>>>(
      attn, u_bfT, nullptr, u_hat, nullptr, GG, HH, GG, GG, TT, HH,
      (long long)GG * GG, (long long)HH * TT, GG, 16, (long long)GG * HH, 1.f);

  // 7) linear attention: lin_kv^T = (lin_k^T @ v / T)^T stored [N=H, K=QKD]
  wmma_gemm<5><<<dim3(HH / 64, QKD / 128, BB), blk, 0, stream>>>(
      lkT, v_bfT, nullptr, nullptr, lin_kvT, QKD, HH, TT, TT, TT, QKD,
      (long long)QKD * TT, (long long)HH * TT, 0, 1, (long long)HH * QKD, 1.f / (float)TT);
  wmma_gemm<5><<<dim3(HH / 64, QKD / 128, BB), blk, 0, stream>>>(
      lkT, u_bfT, nullptr, nullptr, lin_kuT, QKD, HH, TT, TT, TT, QKD,
      (long long)QKD * TT, (long long)HH * TT, 0, 1, (long long)HH * QKD, 1.f / (float)TT);

  // 8) v_hat += lin_q @ lin_kv ;  u_hat += lin_q @ lin_ku   (per batch)
  wmma_gemm<4><<<dim3(HH / 64, TT / 128, BB), blk, 0, stream>>>(
      lqh, lin_kvT, nullptr, v_hat, nullptr, TT, HH, QKD, QKD, QKD, HH,
      (long long)TT * QKD, (long long)HH * QKD, 0, 1, (long long)TT * HH, 1.f);
  wmma_gemm<4><<<dim3(HH / 64, TT / 128, BB), blk, 0, stream>>>(
      lqh, lin_kuT, nullptr, u_hat, nullptr, TT, HH, QKD, QKD, QKD, HH,
      (long long)TT * QKD, (long long)HH * QKD, 0, 1, (long long)TT * HH, 1.f);

  // 9) gating -> gated (reuse h_tmp)
  float* gated = h_tmp;
  gating<<<grid1d((long long)MM * HH), blk, 0, stream>>>(
      u_f, v_f, u_hat, v_hat, gated, (long long)MM * HH);

  // 10) out conv-module: LN -> GEMM(+bias,SiLU) -> dwconv residual, + x
  ln_stats<<<MM, blk, 0, stream>>>(gated, stats, HH);
  ln_apply<<<grid1d((long long)MM * HH), blk, 0, stream>>>(
      gated, stats, o_ln_g, o_ln_b, xn_out, HH, (long long)MM * HH);
  wmma_gemm<1><<<dim3(DD / 64, MM / 128, 1), blk, 0, stream>>>(
      xn_out, Wt_o, o_b, h_out, nullptr, MM, DD, HH, HH, HH, DD, 0, 0, 0, 1, 0, 1.f);
  dwconv<<<grid1d((long long)MM * DD), blk, 0, stream>>>(
      h_out, o_cw, o_cb, x, out, TT, DD, (long long)MM * DD);
}
